// Encoder_48842368090438
// MI455X (gfx1250) — matrix operand order repaired and checked by reading, not compiled, not run on hardware
//
#include <hip/hip_runtime.h>
#include <hip/hip_bf16.h>
#include <math.h>

typedef __attribute__((ext_vector_type(16))) _Float16 v16h;
typedef __attribute__((ext_vector_type(8)))  float    v8f;

#define B_  512
#define T_  128
#define F_  256
#define H_  128
#define G_  512   /* 4*H */

__device__ __forceinline__ float sigmoidf_(float x){ return 1.0f/(1.0f+__expf(-x)); }

// 16-bit A-matrix 16x32 fragment: K offset for f16 pair p (0..7) given lane-half base
__device__ __forceinline__ int afrag_k(int p, int khalf){
    return 2*khalf + 2*p;
}

// Flat shared-aperture pointers carry the wave-relative LDS byte offset in addr[31:0]
__device__ __forceinline__ unsigned lds_off_of(const void* p){
    return (unsigned)(unsigned long long)p;
}
// gfx1250 async copy global->LDS (ASYNCcnt tracked); GV addressing, VDST = LDS offset
__device__ __forceinline__ void async_load_lds_b128(unsigned lds_off, const void* gptr){
    asm volatile("global_load_async_to_lds_b128 %0, %1, off"
                 :: "v"(lds_off), "v"(gptr) : "memory");
}
__device__ __forceinline__ void wait_asynccnt0(){
    asm volatile("s_wait_asynccnt 0x0" ::: "memory");
}

// ---------------- Phase 0: convert weights to f16, fuse biases -------------
__global__ void prep_weights(const float* __restrict__ Wh, const float* __restrict__ Wih,
                             const float* __restrict__ Whh,
                             const float* __restrict__ bih, const float* __restrict__ bhh,
                             _Float16* __restrict__ Wh16, _Float16* __restrict__ Wih16,
                             _Float16* __restrict__ Whh16, float* __restrict__ biasg){
    int i = blockIdx.x*blockDim.x + threadIdx.x;
    if (i < H_*F_) Wh16[i]  = (_Float16)Wh[i];
    if (i < G_*F_) Wih16[i] = (_Float16)Wih[i];
    if (i < G_*H_) Whh16[i] = (_Float16)Whh[i];
    if (i < G_)    biasg[i] = bih[i] + bhh[i];
}

// ---------------- Phase 1: per-(b,f) time recurrences (bandwidth bound) ----
__global__ void prep_series(const float* __restrict__ X, const float* __restrict__ Xlast,
                            const float* __restrict__ Wx, const float* __restrict__ bx,
                            const float* __restrict__ Wattn,
                            _Float16* __restrict__ dX16, float* __restrict__ X0out,
                            float* __restrict__ Sxout, float* __restrict__ cmaskout){
    int gid = blockIdx.x*blockDim.x + threadIdx.x;
    if (gid >= B_*F_) return;
    int b = gid / F_, f = gid % F_;
    const size_t base = (size_t)b*T_*F_ + f;
    float d = 0.f, cnt = 0.f, xsum = 0.f, mprev = 0.f;
    for (int t=0; t<T_; ++t){
        float x  = X[base + (size_t)t*F_];
        float m  = (x != -1.0f) ? 1.f : 0.f;
        float x0 = (m > 0.f) ? x : 0.f;
        if (t > 0) d = (mprev > 0.f) ? 1.f : (d + 1.f);
        dX16[base + (size_t)t*F_]  = (_Float16)d;    // small ints: exact in f16
        X0out[base + (size_t)t*F_] = x0;             // staged in X_tilde output region
        cnt += m; xsum += x0; mprev = m;
    }
    float xmean = (cnt > 0.f) ? (xsum / fmaxf(cnt, 1.f)) : 0.f;
    float wxd = Wx[(size_t)f*F_ + f];
    float bxf = bx[f];
    float sx = 0.f, cm = 0.f;
    for (int t=0; t<T_; ++t){
        float x  = X[base + (size_t)t*F_];
        float m  = (x != -1.0f) ? 1.f : 0.f;
        float x0 = (m > 0.f) ? x : 0.f;
        float dt = (float)dX16[base + (size_t)t*F_];
        float dx = __expf(-fmaxf(dt*wxd + bxf, 0.f));
        float xl = Xlast[base + (size_t)t*F_];
        float xnew = (1.f-m)*(dx*xl + (1.f-dx)*xmean) + m*x0;
        sx += xnew * Wattn[2*H_ + t];
        Sxout[base + (size_t)t*F_] = sx;
        cm += m * Wattn[2*H_ + T_ + t];
    }
    cmaskout[(size_t)b*F_ + f] = cm;
}

// ---------------- Phase 2: Dh = exp(-relu(deltaX @ W_h^T + b_h)) via WMMA --
// One 16x16 tile per wave; output stored f16 (values in (0,1]) to halve traffic.
__global__ __launch_bounds__(256) void dh_gemm(const _Float16* __restrict__ dX16,
                                               const _Float16* __restrict__ Wh16,
                                               const float* __restrict__ bh,
                                               _Float16* __restrict__ Dh16){
    const int l     = threadIdx.x & 31;
    const int job   = blockIdx.x*8 + (threadIdx.x >> 5);
    const int mtile = job >> 3;
    const int ntile = job & 7;
    const int m     = l & 15;
    const int khalf = (l >> 4) * 8;
    const int bk    = (l >> 4) * 16;
    const int n     = ntile*16 + (l & 15);
    const _Float16* arow = dX16 + (size_t)(mtile*16 + m) * F_;
    const _Float16* brow = Wh16 + (size_t)n * F_;     // B[k][n] = W_h[n][k], k contiguous
    v8f acc = {};
    #pragma unroll
    for (int kk=0; kk<8; ++kk){
        const int kb = kk*32;
        v16h a;
        #pragma unroll
        for (int p=0; p<8; ++p){
            int k0 = kb + afrag_k(p, khalf);
            a[2*p]   = arow[k0];
            a[2*p+1] = arow[k0+1];
        }
        v16h bf = *(const v16h*)(brow + kb + bk);
        acc = __builtin_amdgcn_wmma_f32_16x16x32_f16(false, a, false, bf, (short)0, acc, false, false);
    }
    float bn = bh[n];
    #pragma unroll
    for (int r=0; r<8; ++r){
        int rowM = r + 8*(l >> 4);
        Dh16[(size_t)(mtile*16 + rowM)*H_ + n] = (_Float16)__expf(-fmaxf(acc[r] + bn, 0.f));
    }
}

// ---------------- Phase 3: sequential attention-LSTM scan ------------------
// 32 persistent workgroups, 16 batch rows each; h/c resident in LDS;
// Dh_{t+1} prefetched via global_load_async_to_lds_b128 overlapping the GEMM.
__global__ __launch_bounds__(256) void scan_kernel(
    const float* __restrict__ Wattn, const float* __restrict__ battn,
    const _Float16* __restrict__ Wih16, const _Float16* __restrict__ Whh16,
    const float* __restrict__ biasg,
    const float* __restrict__ Sx, const float* __restrict__ cmask,
    const _Float16* __restrict__ Dh16,
    float* __restrict__ Xtilde /* holds X0 on entry */, float* __restrict__ Xenc){

    __shared__ float    h32[16*H_];       // 8 KB
    __shared__ float    c32[16*H_];       // 8 KB
    __shared__ _Float16 h16[16*H_];       // 4 KB
    __shared__ _Float16 xh[16*F_];        // 8 KB
    __shared__ float    gates[16*G_];     // 32 KB (reductions aliased into it)
    __shared__ _Float16 dh_st[16*H_];     // 4 KB async staging   => 64 KB total

    // temporally-disjoint aliases (used only before GEMM writes gates)
    float* red  = gates;          // [0,256)
    float* srow = gates + 256;    // [256,272)
    float* rmax = gates + 272;    // [272,288)
    float* rsum = gates + 288;    // [288,304)

    const int tid  = threadIdx.x;
    const int l    = tid & 31;
    const int w    = tid >> 5;
    const int b0   = blockIdx.x * 16;
    const int row  = tid >> 4;     // 16 threads per batch row
    const int seg  = tid & 15;
    const int fbase = seg * 16;    // 16 F-columns per thread
    const int j8   = seg * 8;      // 8 H-columns per thread

    for (int i=tid; i<16*H_; i+=256){ h32[i]=0.f; c32[i]=0.f; h16[i]=(_Float16)0.f; }

    float cmreg[16];
    #pragma unroll
    for (int j=0; j<16; ++j) cmreg[j] = cmask[(size_t)(b0+row)*F_ + fbase + j];
    float wah[8], was[8];
    #pragma unroll
    for (int k=0; k<8; ++k){ wah[k]=Wattn[j8+k]; was[k]=Wattn[H_+j8+k]; }
    const float ba = battn[0];

    // async staging setup: each thread copies 16B of the 16x128 f16 Dh tile
    const _Float16* dh_src0 = Dh16 + (size_t)(b0+row)*T_*H_ + seg*8;
    const unsigned  dh_dst  = lds_off_of(&dh_st[row*H_ + seg*8]);
    async_load_lds_b128(dh_dst, dh_src0);        // stage t=0
    wait_asynccnt0();
    __syncthreads();

    for (int t=0; t<T_; ++t){
        // h = Dh_t * h (pre-score, per reference), refresh f16 mirror
        for (int i=tid; i<16*H_; i+=256){
            int r = i >> 7, j = i & (H_-1);
            float hv = h32[i] * (float)dh_st[r*H_ + j];
            h32[i] = hv; h16[i] = (_Float16)hv;
        }
        __syncthreads();
        // Dh_t consumed: prefetch Dh_{t+1} into LDS, overlapped with softmax+GEMM
        if (t+1 < T_)
            async_load_lds_b128(dh_dst, dh_src0 + (size_t)(t+1)*H_);
        // per-row scalar: h.wa_h + c.wa_s + b_attn
        {
            float part = 0.f;
            #pragma unroll
            for (int k=0; k<8; ++k)
                part += h32[row*H_ + j8 + k]*wah[k] + c32[row*H_ + j8 + k]*was[k];
            red[tid] = part;
        }
        __syncthreads();
        if (seg == 0){
            float s = ba;
            #pragma unroll
            for (int k=0; k<16; ++k) s += red[row*16+k];
            srow[row] = s;
        }
        __syncthreads();
        // score + softmax over F=256 (16 threads/row, 16 cols/thread)
        float sc[16];
        {
            const float* sxr = Sx + ((size_t)(b0+row)*T_ + t)*F_ + fbase;
            float s0 = srow[row], mx = -3.4e38f;
            #pragma unroll
            for (int j=0; j<16; ++j){ float v = s0 + sxr[j] + cmreg[j]; sc[j]=v; mx=fmaxf(mx,v); }
            red[tid] = mx;
        }
        __syncthreads();
        if (seg == 0){ float m2=-3.4e38f; for (int k=0;k<16;++k) m2=fmaxf(m2,red[row*16+k]); rmax[row]=m2; }
        __syncthreads();
        {
            float rm = rmax[row], ps = 0.f;
            #pragma unroll
            for (int j=0; j<16; ++j){ float e=__expf(sc[j]-rm); sc[j]=e; ps+=e; }
            red[tid] = ps;
        }
        __syncthreads();
        if (seg == 0){ float s=0.f; for (int k=0;k<16;++k) s+=red[row*16+k]; rsum[row]=s; }
        __syncthreads();
        {
            float inv = 1.f / rsum[row];
            float* xtr = Xtilde + ((size_t)(b0+row)*T_ + t)*F_ + fbase;
            #pragma unroll
            for (int j=0; j<16; ++j){
                float xt = sc[j]*inv * xtr[j];     // xtr currently holds X0
                xtr[j] = xt;                       // emit X_tilde
                xh[row*F_ + fbase + j] = (_Float16)xt;
            }
        }
        __syncthreads();
        // gates = x_t @ W_ih^T + h @ W_hh^T  (f16 WMMA, f32 accumulate)
        {
            const int m     = l & 15;
            const int khalf = (l >> 4) * 8;
            const int bk    = (l >> 4) * 16;
            #pragma unroll
            for (int q=0; q<4; ++q){
                const int nt = w*4 + q;
                const int n  = nt*16 + (l & 15);
                v8f acc = {};
                const _Float16* brow = Wih16 + (size_t)n*F_;
                #pragma unroll
                for (int kk=0; kk<8; ++kk){
                    const int kb = kk*32;
                    v16h a;
                    #pragma unroll
                    for (int p=0; p<8; ++p){
                        int k0 = kb + afrag_k(p, khalf);
                        a[2*p]   = xh[m*F_ + k0];
                        a[2*p+1] = xh[m*F_ + k0 + 1];
                    }
                    v16h bf = *(const v16h*)(brow + kb + bk);
                    acc = __builtin_amdgcn_wmma_f32_16x16x32_f16(false,a,false,bf,(short)0,acc,false,false);
                }
                const _Float16* brow2 = Whh16 + (size_t)n*H_;
                #pragma unroll
                for (int kk=0; kk<4; ++kk){
                    const int kb = kk*32;
                    v16h a;
                    #pragma unroll
                    for (int p=0; p<8; ++p){
                        int k0 = kb + afrag_k(p, khalf);
                        a[2*p]   = h16[m*H_ + k0];
                        a[2*p+1] = h16[m*H_ + k0 + 1];
                    }
                    v16h bf = *(const v16h*)(brow2 + kb + bk);
                    acc = __builtin_amdgcn_wmma_f32_16x16x32_f16(false,a,false,bf,(short)0,acc,false,false);
                }
                #pragma unroll
                for (int r=0; r<8; ++r)
                    gates[(r + 8*(l>>4))*G_ + n] = acc[r];
            }
        }
        __syncthreads();
        // LSTM pointwise update + emit X_enc
        for (int i=tid; i<16*H_; i+=256){
            int r = i >> 7, j = i & (H_-1);
            float ii = gates[r*G_ + j]        + biasg[j];
            float ff = gates[r*G_ + H_ + j]   + biasg[H_+j];
            float gg = gates[r*G_ + 2*H_ + j] + biasg[2*H_+j];
            float oo = gates[r*G_ + 3*H_ + j] + biasg[3*H_+j];
            float cv = sigmoidf_(ff)*c32[i] + sigmoidf_(ii)*tanhf(gg);
            float hv = sigmoidf_(oo)*tanhf(cv);
            c32[i]=cv; h32[i]=hv; h16[i]=(_Float16)hv;
            Xenc[((size_t)(b0+r)*T_ + t)*H_ + j] = hv;
        }
        if (t+1 < T_)   // L2 prefetch for next step's Sx row
            __builtin_prefetch(Sx + ((size_t)(b0+row)*T_ + (t+1))*F_ + fbase, 0, 1);
        wait_asynccnt0();           // Dh_{t+1} staging complete
        __syncthreads();
    }
}

extern "C" void kernel_launch(void* const* d_in, const int* in_sizes, int n_in,
                              void* d_out, int out_size, void* d_ws, size_t ws_size,
                              hipStream_t stream){
    (void)in_sizes; (void)n_in; (void)out_size; (void)ws_size;
    const float* X     = (const float*)d_in[0];
    const float* Xlast = (const float*)d_in[1];
    const float* Wx    = (const float*)d_in[2];
    const float* bx    = (const float*)d_in[3];
    const float* Wh    = (const float*)d_in[4];
    const float* bh    = (const float*)d_in[5];
    const float* Wih   = (const float*)d_in[6];
    const float* bih   = (const float*)d_in[7];
    const float* Whh   = (const float*)d_in[8];
    const float* bhh   = (const float*)d_in[9];
    const float* Wattn = (const float*)d_in[10];
    const float* battn = (const float*)d_in[11];

    // Workspace carve (~118 MB total), 256B-aligned regions
    char* ws = (char*)d_ws;
    size_t off = 0;
    auto carve = [&](size_t bytes)->char*{
        char* p = ws + off;
        off = (off + bytes + 255) & ~(size_t)255;
        return p;
    };
    _Float16* dX16   = (_Float16*)carve((size_t)B_*T_*F_*sizeof(_Float16));
    _Float16* Wh16   = (_Float16*)carve((size_t)H_*F_*sizeof(_Float16));
    _Float16* Wih16  = (_Float16*)carve((size_t)G_*F_*sizeof(_Float16));
    _Float16* Whh16  = (_Float16*)carve((size_t)G_*H_*sizeof(_Float16));
    float*    biasg  = (float*)   carve((size_t)G_*sizeof(float));
    float*    Sx     = (float*)   carve((size_t)B_*T_*F_*sizeof(float));
    float*    cmaskp = (float*)   carve((size_t)B_*F_*sizeof(float));
    _Float16* Dh16   = (_Float16*)carve((size_t)B_*T_*H_*sizeof(_Float16));

    float* Xtilde = (float*)d_out;                 // (B,T,F); staged with X0 first
    float* Xenc   = Xtilde + (size_t)B_*T_*F_;     // (B,T,H)

    prep_weights<<<(G_*F_ + 255)/256, 256, 0, stream>>>(Wh, Wih, Whh, bih, bhh,
                                                        Wh16, Wih16, Whh16, biasg);
    prep_series<<<(B_*F_ + 255)/256, 256, 0, stream>>>(X, Xlast, Wx, bx, Wattn,
                                                       dX16, Xtilde, Sx, cmaskp);
    dh_gemm<<<(B_*T_/16), 256, 0, stream>>>(dX16, Wh16, bh, Dh16);
    scan_kernel<<<B_/16, 256, 0, stream>>>(Wattn, battn, Wih16, Whh16, biasg,
                                           Sx, cmaskp, Dh16, Xtilde, Xenc);
}
